// RelationalNetwork_73890617360507
// MI455X (gfx1250) — matrix operand order repaired and checked by reading, not compiled, not run on hardware
//
#include <hip/hip_runtime.h>
#include <hip/hip_bf16.h>

typedef __attribute__((ext_vector_type(16))) _Float16 v16h;
typedef __attribute__((ext_vector_type(8)))  _Float16 v8h;
typedef __attribute__((ext_vector_type(8)))  float    v8f;

#define Bb   64
#define Ll   32
#define Ee   32
#define Hh   128
#define Kk   24
#define Gg   256
#define Ff   256
#define NCc  28
#define NOBJ 64
#define C2H  176   // 2*K + H
#define ROWS 64    // pair-rows per workgroup tile

// ---------------- helpers ----------------
union V16U { v16h v; v8h h[2]; };
__device__ __forceinline__ v16h cat8(v8h lo, v8h hi) {
  V16U u; u.h[0] = lo; u.h[1] = hi; return u.v;
}

// ---------------- conv + BN + LeakyReLU ----------------
__global__ void conv_s2(const float* __restrict__ x, const float* __restrict__ w,
                        const float* __restrict__ bias, float* __restrict__ z,
                        int cin, int win, int wout) {
  int idx = blockIdx.x * blockDim.x + threadIdx.x;
  int total = Bb * Kk * wout * wout;
  if (idx >= total) return;
  int xo = idx % wout; int t = idx / wout;
  int yo = t % wout;   t /= wout;
  int oc = t % Kk;     int b = t / Kk;
  float acc = bias[oc];
  int yi0 = 2 * yo - 1, xi0 = 2 * xo - 1;
  for (int ic = 0; ic < cin; ++ic) {
    const float* xp = x + ((size_t)(b * cin + ic)) * win * win;
    const float* wp = w + ((oc * cin + ic) * 9);
    #pragma unroll
    for (int ky = 0; ky < 3; ++ky) {
      int yi = yi0 + ky;
      if (yi < 0 || yi >= win) continue;
      #pragma unroll
      for (int kx = 0; kx < 3; ++kx) {
        int xi = xi0 + kx;
        if (xi < 0 || xi >= win) continue;
        acc += xp[yi * win + xi] * wp[ky * 3 + kx];
      }
    }
  }
  z[idx] = acc;
}

__global__ __launch_bounds__(256) void bn_stats(const float* __restrict__ z,
                                                float* __restrict__ stats, int wout) {
  int c = blockIdx.x;            // one block per channel
  int n = wout * wout;
  float s = 0.f, s2 = 0.f;
  for (int b = 0; b < Bb; ++b) {
    const float* p = z + ((size_t)(b * Kk + c)) * n;
    for (int i = threadIdx.x; i < n; i += 256) { float v = p[i]; s += v; s2 += v * v; }
  }
  __shared__ float sh[512];
  sh[threadIdx.x] = s; sh[256 + threadIdx.x] = s2;
  __syncthreads();
  for (int st = 128; st > 0; st >>= 1) {
    if (threadIdx.x < st) {
      sh[threadIdx.x]       += sh[threadIdx.x + st];
      sh[256 + threadIdx.x] += sh[256 + threadIdx.x + st];
    }
    __syncthreads();
  }
  if (threadIdx.x == 0) {
    float cnt = (float)(Bb * n);
    float mu  = sh[0] / cnt;
    float var = sh[256] / cnt - mu * mu;          // biased variance (training BN)
    stats[2 * c]     = mu;
    stats[2 * c + 1] = rsqrtf(var + 1e-5f);
  }
}

__global__ void bn_apply(float* __restrict__ z, const float* __restrict__ stats,
                         const float* __restrict__ g, const float* __restrict__ beta,
                         int wout) {
  int idx = blockIdx.x * blockDim.x + threadIdx.x;
  int n = wout * wout;
  int total = Bb * Kk * n;
  if (idx >= total) return;
  int c = (idx / n) % Kk;
  float y = (z[idx] - stats[2 * c]) * stats[2 * c + 1] * g[c] + beta[c];
  z[idx] = y >= 0.f ? y : 0.1f * y;               // LeakyReLU(0.1)
}

// ---------------- embedding + masked LSTM ----------------
__global__ __launch_bounds__(128) void embed_lstm(
    const int* __restrict__ q, const int* __restrict__ qlen,
    const float* __restrict__ emb,
    const float* __restrict__ wih, const float* __restrict__ whh,
    const float* __restrict__ bih, const float* __restrict__ bhh,
    float* __restrict__ qout) {
  int b = blockIdx.x, tid = threadIdx.x;
  __shared__ float xh[Ee + Hh];
  __shared__ float hs[Hh], cs[Hh], gates[4 * Hh];
  if (tid < Hh) { hs[tid] = 0.f; cs[tid] = 0.f; }
  __syncthreads();
  int len = qlen[b];                               // uniform per block; mask is monotone
  for (int t = 0; t < Ll && t < len; ++t) {
    int tok = q[b * Ll + t];
    if (tid < Ee) xh[tid] = emb[tok * Ee + tid];
    if (tid < Hh) xh[Ee + tid] = hs[tid];
    __syncthreads();
    for (int gi = tid; gi < 4 * Hh; gi += 128) {
      float acc = bih[gi] + bhh[gi];
      const float* wi = wih + (size_t)gi * Ee;
      for (int e = 0; e < Ee; ++e) acc += wi[e] * xh[e];
      const float* wh = whh + (size_t)gi * Hh;
      for (int h = 0; h < Hh; ++h) acc += wh[h] * xh[Ee + h];
      gates[gi] = acc;
    }
    __syncthreads();
    if (tid < Hh) {                                 // PyTorch gate order: i,f,g,o
      float ig = 1.f / (1.f + expf(-gates[tid]));
      float fg = 1.f / (1.f + expf(-gates[Hh + tid]));
      float gg = tanhf(gates[2 * Hh + tid]);
      float og = 1.f / (1.f + expf(-gates[3 * Hh + tid]));
      float cn = fg * cs[tid] + ig * gg;
      cs[tid] = cn;
      hs[tid] = og * tanhf(cn);
    }
    __syncthreads();
  }
  if (tid < Hh) qout[b * Hh + tid] = hs[tid];
}

// ---------------- g-layer-1 rank decomposition ----------------
// Ao = obj @ W1[:, :24]^T ; Bo = obj @ W1[:, 24:48]^T ; Qc = q @ W1[:, 48:]^T + b1
__global__ __launch_bounds__(256) void g0_pre(
    const float* __restrict__ convout /*[B,24,8,8]*/, const float* __restrict__ qv,
    const float* __restrict__ w1 /*[256,176]*/, const float* __restrict__ b1,
    float* __restrict__ Ao, float* __restrict__ Bo, float* __restrict__ Qc) {
  int b = blockIdx.x, n = threadIdx.x;
  __shared__ float obj[NOBJ][Kk];
  for (int idx = n; idx < NOBJ * Kk; idx += 256) {
    int o = idx / Kk, c = idx % Kk;
    obj[o][c] = convout[(((size_t)b * Kk + c) * 8 + (o >> 3)) * 8 + (o & 7)];
  }
  __syncthreads();
  float accq = b1[n];
  for (int h = 0; h < Hh; ++h) accq += w1[n * C2H + 2 * Kk + h] * qv[b * Hh + h];
  Qc[b * Gg + n] = accq;
  for (int o = 0; o < NOBJ; ++o) {
    float a = 0.f, bb = 0.f;
    #pragma unroll
    for (int c = 0; c < Kk; ++c) {
      a  += w1[n * C2H + c]      * obj[o][c];
      bb += w1[n * C2H + Kk + c] * obj[o][c];
    }
    Ao[((size_t)b * NOBJ + o) * Gg + n] = a;
    Bo[((size_t)b * NOBJ + o) * Gg + n] = bb;
  }
}

// ---------------- misc ----------------
__global__ void cvt_f16(const float* __restrict__ s, _Float16* __restrict__ d, int n) {
  int i = blockIdx.x * blockDim.x + threadIdx.x;
  if (i < n) d[i] = (_Float16)s[i];
}
__global__ void zero_f32(float* __restrict__ p, int n) {
  int i = blockIdx.x * blockDim.x + threadIdx.x;
  if (i < n) p[i] = 0.f;
}

// ---------------- fused g-network (WMMA) ----------------
// One 16-row strip x 8 col-tiles per wave; waves 0-3 cols [0,128), waves 4-7 cols [128,256).
__device__ __forceinline__ void gemm_layer(
    const _Float16 (*__restrict__ actIn)[Gg], _Float16 (*__restrict__ actOut)[Gg],
    const _Float16* __restrict__ W /*[256x256 row-major, (out,in)]*/,
    const float* __restrict__ bias, int wave, int lane) {
  const int rs = (wave & 3) * 16;
  const int nh = wave >> 2;
  const int lg = lane >> 4;     // half-wave group
  const int ll = lane & 15;
  const v8f vz = {0.f, 0.f, 0.f, 0.f, 0.f, 0.f, 0.f, 0.f};
  v8f acc[8];
  #pragma unroll
  for (int t = 0; t < 8; ++t) acc[t] = vz;

  for (int ks = 0; ks < 8; ++ks) {
    const int k0 = ks * 32;
    // A-tile (16x32 f16): lane-group g holds K in [16g,16g+16)
    const int kbA = k0 + lg * 16;
    v8h alo = *(const v8h*)&actIn[rs + ll][kbA];
    v8h ahi = *(const v8h*)&actIn[rs + ll][kbA + 8];
    v16h a = cat8(alo, ahi);
    #pragma unroll
    for (int ct = 0; ct < 8; ++ct) {
      const int n0 = nh * 128 + ct * 16;
      // B-tile (32x16): column n = row n of W; lane-group g holds K in [16g,16g+16)
      v16h bm = *(const v16h*)(W + (size_t)(n0 + ll) * Gg + k0 + lg * 16);
      acc[ct] = __builtin_amdgcn_wmma_f32_16x16x32_f16(
          false, a, false, bm, (short)0, acc[ct], false, false);
    }
  }
  // bias + ReLU + write f16 to LDS (C layout: VGPR r -> row rs + 8*lg + r, col = n0+ll)
  #pragma unroll
  for (int ct = 0; ct < 8; ++ct) {
    const int col = nh * 128 + ct * 16 + ll;
    const float bv = bias[col];
    const int rbase = rs + lg * 8;
    #pragma unroll
    for (int r = 0; r < 8; ++r) {
      float v = acc[ct][r] + bv;
      v = v > 0.f ? v : 0.f;
      actOut[rbase + r][col] = (_Float16)v;
    }
  }
  __syncthreads();
}

__global__ __launch_bounds__(256) void rn_g_fused(
    const float* __restrict__ Ao, const float* __restrict__ Bo, const float* __restrict__ Qc,
    const _Float16* __restrict__ W1, const float* __restrict__ B1,
    const _Float16* __restrict__ W2, const float* __restrict__ B2,
    const _Float16* __restrict__ W3, const float* __restrict__ B3,
    float* __restrict__ gsum) {
  __shared__ _Float16 act0[ROWS][Gg];   // 32 KB
  __shared__ _Float16 act1[ROWS][Gg];   // 32 KB
  const int b = blockIdx.x;
  const int pbase = blockIdx.y * ROWS;
  const int tid = threadIdx.x;

  // layer 1: h1 = relu(Ao[i] + Bo[j] + Qc) built straight into LDS as f16
  for (int idx = tid; idx < ROWS * Gg; idx += 256) {
    const int r = idx >> 8, n = idx & 255;
    const int p = pbase + r, i = p >> 6, j = p & 63;
    float v = Ao[((size_t)b * NOBJ + i) * Gg + n]
            + Bo[((size_t)b * NOBJ + j) * Gg + n]
            + Qc[b * Gg + n];
    act0[r][n] = (_Float16)(v > 0.f ? v : 0.f);
  }
  __syncthreads();

  const int wave = tid >> 5, lane = tid & 31;
  gemm_layer(act0, act1, W1, B1, wave, lane);   // g layer 2
  gemm_layer(act1, act0, W2, B2, wave, lane);   // g layer 3
  gemm_layer(act0, act1, W3, B3, wave, lane);   // g layer 4

  // sum over pairs in this tile -> global accumulator
  float s = 0.f;
  for (int r = 0; r < ROWS; ++r) s += (float)act1[r][tid];
  atomicAdd(&gsum[b * Gg + tid], s);
}

// ---------------- f-MLP ----------------
__global__ __launch_bounds__(256) void f_mlp(
    const float* __restrict__ gsum,
    const float* __restrict__ f0w, const float* __restrict__ f0b,
    const float* __restrict__ f1w, const float* __restrict__ f1b,
    const float* __restrict__ f2w, const float* __restrict__ f2b,
    float* __restrict__ out) {
  int b = blockIdx.x, n = threadIdx.x;
  __shared__ float xin[Gg], y0[Ff], y1[Ff];
  xin[n] = gsum[b * Gg + n];
  __syncthreads();
  float acc = f0b[n];
  for (int k = 0; k < Gg; ++k) acc += f0w[(size_t)n * Gg + k] * xin[k];
  y0[n] = acc > 0.f ? acc : 0.f;
  __syncthreads();
  acc = f1b[n];
  for (int k = 0; k < Ff; ++k) acc += f1w[(size_t)n * Ff + k] * y0[k];
  y1[n] = acc > 0.f ? acc : 0.f;
  __syncthreads();
  if (n < NCc) {
    acc = f2b[n];
    for (int k = 0; k < Ff; ++k) acc += f2w[(size_t)n * Ff + k] * y1[k];
    out[b * NCc + n] = acc;
  }
}

// ---------------- host ----------------
extern "C" void kernel_launch(void* const* d_in, const int* in_sizes, int n_in,
                              void* d_out, int out_size, void* d_ws, size_t ws_size,
                              hipStream_t stream) {
  (void)in_sizes; (void)n_in; (void)out_size; (void)ws_size;
  const float* image    = (const float*)d_in[0];
  const int*   question = (const int*)d_in[1];
  const int*   qlen     = (const int*)d_in[2];
  const float *cw[4], *cb[4], *cg[4], *cbe[4];
  for (int l = 0; l < 4; ++l) {
    cw[l]  = (const float*)d_in[3 + 4 * l];
    cb[l]  = (const float*)d_in[4 + 4 * l];
    cg[l]  = (const float*)d_in[5 + 4 * l];
    cbe[l] = (const float*)d_in[6 + 4 * l];
  }
  const float* emb = (const float*)d_in[19];
  const float* wih = (const float*)d_in[20];
  const float* whh = (const float*)d_in[21];
  const float* bih = (const float*)d_in[22];
  const float* bhh = (const float*)d_in[23];
  const float* g0w = (const float*)d_in[24]; const float* g0b = (const float*)d_in[25];
  const float* g1w = (const float*)d_in[26]; const float* g1b = (const float*)d_in[27];
  const float* g2w = (const float*)d_in[28]; const float* g2b = (const float*)d_in[29];
  const float* g3w = (const float*)d_in[30]; const float* g3b = (const float*)d_in[31];
  const float* f0w = (const float*)d_in[32]; const float* f0b = (const float*)d_in[33];
  const float* f1w = (const float*)d_in[34]; const float* f1b = (const float*)d_in[35];
  const float* f2w = (const float*)d_in[36]; const float* f2b = (const float*)d_in[37];

  char* ws = (char*)d_ws; size_t off = 0;
  auto carve = [&](size_t bytes) -> void* {
    void* p = ws + off; off += (bytes + 255) & ~(size_t)255; return p;
  };
  float* z1    = (float*)carve((size_t)Bb * Kk * 64 * 64 * 4);
  float* z2    = (float*)carve((size_t)Bb * Kk * 32 * 32 * 4);
  float* z3    = (float*)carve((size_t)Bb * Kk * 16 * 16 * 4);
  float* z4    = (float*)carve((size_t)Bb * Kk * 8 * 8 * 4);
  float* stats = (float*)carve(2 * Kk * 4);
  float* qh    = (float*)carve((size_t)Bb * Hh * 4);
  float* Ao    = (float*)carve((size_t)Bb * NOBJ * Gg * 4);
  float* Bo    = (float*)carve((size_t)Bb * NOBJ * Gg * 4);
  float* Qc    = (float*)carve((size_t)Bb * Gg * 4);
  float* gsum  = (float*)carve((size_t)Bb * Gg * 4);
  _Float16* g1h = (_Float16*)carve((size_t)Gg * Gg * 2);
  _Float16* g2h = (_Float16*)carve((size_t)Gg * Gg * 2);
  _Float16* g3h = (_Float16*)carve((size_t)Gg * Gg * 2);

  // weights -> f16 for WMMA
  cvt_f16<<<(Gg * Gg + 255) / 256, 256, 0, stream>>>(g1w, g1h, Gg * Gg);
  cvt_f16<<<(Gg * Gg + 255) / 256, 256, 0, stream>>>(g2w, g2h, Gg * Gg);
  cvt_f16<<<(Gg * Gg + 255) / 256, 256, 0, stream>>>(g3w, g3h, Gg * Gg);

  // conv tower: 128 -> 64 -> 32 -> 16 -> 8
  const float* src = image;
  float* bufs[4] = {z1, z2, z3, z4};
  int win = 128, cin = 3;
  for (int l = 0; l < 4; ++l) {
    int wout = win / 2;
    int total = Bb * Kk * wout * wout;
    conv_s2<<<(total + 255) / 256, 256, 0, stream>>>(src, cw[l], cb[l], bufs[l], cin, win, wout);
    bn_stats<<<Kk, 256, 0, stream>>>(bufs[l], stats, wout);
    bn_apply<<<(total + 255) / 256, 256, 0, stream>>>(bufs[l], stats, cg[l], cbe[l], wout);
    src = bufs[l]; win = wout; cin = Kk;
  }

  // question path
  embed_lstm<<<Bb, 128, 0, stream>>>(question, qlen, emb, wih, whh, bih, bhh, qh);

  // g layer-1 decomposition
  g0_pre<<<Bb, 256, 0, stream>>>(z4, qh, g0w, g0b, Ao, Bo, Qc);

  // fused g-network over all pairs
  zero_f32<<<(Bb * Gg + 255) / 256, 256, 0, stream>>>(gsum, Bb * Gg);
  rn_g_fused<<<dim3(Bb, (NOBJ * NOBJ) / ROWS), 256, 0, stream>>>(
      Ao, Bo, Qc, g1h, g1b, g2h, g2b, g3h, g3b, gsum);

  // f-MLP head
  f_mlp<<<Bb, 256, 0, stream>>>(gsum, f0w, f0b, f1w, f1b, f2w, f2b, (float*)d_out);
}
